// VanillaRNN_16810501996835
// MI455X (gfx1250) — matrix operand order repaired and checked by reading, not compiled, not run on hardware
//
#include <hip/hip_runtime.h>
#include <math.h>

#define V_DIM 32000
#define E_DIM 512
#define H_DIM 1024
#define B_DIM 16
#define T_DIM 256
#define BT_DIM (B_DIM * T_DIM)

typedef __bf16 bf16_t;
typedef bf16_t v16bf __attribute__((ext_vector_type(16)));
typedef bf16_t v8bf  __attribute__((ext_vector_type(8)));
typedef float  v8f   __attribute__((ext_vector_type(8)));

typedef unsigned int tdm_v4u __attribute__((ext_vector_type(4)));
typedef int          tdm_v4i __attribute__((ext_vector_type(4)));
typedef int          tdm_v8i __attribute__((ext_vector_type(8)));

__device__ __forceinline__ bf16_t f2bf(float f) {
    return (bf16_t)f;  // native v_cvt (RNE) on gfx1250
}

__device__ __forceinline__ v8f wmma_bf16(v16bf a, v16bf b, v8f c) {
    // (neg_a, A, neg_b, B, c_mod, C, reuse_a, reuse_b)
    return __builtin_amdgcn_wmma_f32_16x16x32_bf16(false, a, false, b, (short)0, c,
                                                   false, false);
}

// B fragment: 16 contiguous bf16 (K run), 32B aligned.
__device__ __forceinline__ v16bf load_b_frag(const bf16_t* p) {
    return *(const v16bf*)p;
}

// A fragment: elems 0..7 at p[0..7], elems 8..15 at p[8..15] (two 16B loads).
__device__ __forceinline__ v16bf load_a_split(const bf16_t* p) {
    v8bf lo = *(const v8bf*)(p);
    v8bf hi = *(const v8bf*)(p + 8);
    v16bf a;
#pragma unroll
    for (int e = 0; e < 8; ++e) { a[e] = lo[e]; a[e + 8] = hi[e]; }
    return a;
}

// --------- Tensor Data Mover: 2D bf16 tile, global -> LDS -------
// tile_x elems contiguous (dim0), tile_y rows (dim1), global row stride
// gstride_elems. Optional LDS row padding via D# pad fields.
__device__ __forceinline__ void tdm_load_2d_bf16(unsigned lds_off, const void* gptr,
                                                 unsigned tile_x, unsigned tile_y,
                                                 unsigned long long gstride_elems,
                                                 unsigned pad_enable,
                                                 unsigned pad_interval,
                                                 unsigned pad_amount) {
    unsigned long long ga = (unsigned long long)(size_t)gptr;

    tdm_v4u g0;
    g0[0] = 1u;                                     // count=1 valid user D#
    g0[1] = lds_off;                                // lds_addr (bytes)
    g0[2] = (unsigned)ga;                           // global_addr[31:0]
    g0[3] = (unsigned)((ga >> 32) & 0x01FFFFFFull)  // global_addr[56:32]
          | (2u << 30);                             // type = 2 ("image")

    const unsigned long long td0 = 0x40000000ull;   // huge extents: no OOB clip
    const unsigned long long td1 = 0x40000000ull;

    unsigned long long q0 = (1ull << 16)            // data_size = 1 -> 2 bytes
        | ((unsigned long long)(pad_enable & 1u) << 20)
        | ((unsigned long long)(pad_interval & 7u) << 22)
        | ((unsigned long long)(pad_amount & 127u) << 25)
        | ((td0 & 0xFFFFull) << 48);                        // tensor_dim0[15:0]
    unsigned long long q1 = ((td0 >> 16) & 0xFFFFull)       // tensor_dim0[31:16]
        | ((td1 & 0xFFFFFFFFull) << 16)                     // tensor_dim1
        | ((unsigned long long)(tile_x & 0xFFFFu) << 48);   // tile_dim0
    unsigned long long q2 = (unsigned long long)(tile_y & 0xFFFFu)  // tile_dim1
        | ((gstride_elems & 0xFFFFFFFFull) << 32);          // dim0_stride[31:0]
    unsigned long long q3 = (gstride_elems >> 32) & 0xFFFFull;  // dim0_stride[47:32]

    tdm_v8i g1;
    g1[0] = (int)q0; g1[1] = (int)(q0 >> 32);
    g1[2] = (int)q1; g1[3] = (int)(q1 >> 32);
    g1[4] = (int)q2; g1[5] = (int)(q2 >> 32);
    g1[6] = (int)q3; g1[7] = (int)(q3 >> 32);

    tdm_v4i z4 = {0, 0, 0, 0};
#if __clang_major__ >= 23
    tdm_v8i z8 = {0, 0, 0, 0, 0, 0, 0, 0};
    __builtin_amdgcn_tensor_load_to_lds(g0, g1, z4, z4, z8, 0);
#else
    __builtin_amdgcn_tensor_load_to_lds(g0, g1, z4, z4, 0);
#endif
}

// ---------------- fp32 -> bf16 weight conversion ----------------
__global__ void k_f32_to_bf16(const float* __restrict__ src, bf16_t* __restrict__ dst,
                              long n) {
    long i = (long)blockIdx.x * blockDim.x + threadIdx.x;
    long stride = (long)gridDim.x * blockDim.x;
    for (; i < n; i += stride) dst[i] = f2bf(src[i]);
}

// ---------------- embedding gather + input projection -----------
// pre[m, n] = sum_k emb[x[m]][k] * W_ih[n][k] + b_ih[n]
__global__ void k_embed_proj(const int* __restrict__ x, const float* __restrict__ emb,
                             const bf16_t* __restrict__ Wih, const float* __restrict__ b_ih,
                             float* __restrict__ pre) {
    const int lane = threadIdx.x & 31;
    const int wave = threadIdx.x >> 5;
    const int half = lane >> 4;
    const int ln   = lane & 15;
    const int m0   = blockIdx.x * 16;
    const int col0 = blockIdx.y * 128 + wave * 16;

    const int tok = x[m0 + ln];
    const float*  erow = emb + (size_t)tok * E_DIM;
    const bf16_t* wrow = Wih + (size_t)(col0 + ln) * E_DIM;

    v8f acc = {};
    for (int kc = 0; kc < E_DIM; kc += 32) {
        const float* ap = erow + kc + half * 16;
        v8f lo = *(const v8f*)(ap);
        v8f hi = *(const v8f*)(ap + 8);
        v16bf a;
#pragma unroll
        for (int e = 0; e < 8; ++e) { a[e] = f2bf(lo[e]); a[e + 8] = f2bf(hi[e]); }
        v16bf b = load_b_frag(wrow + kc + half * 16);
        acc = wmma_bf16(a, b, acc);
    }
    const int n = col0 + ln;
    const float bias = b_ih[n];
#pragma unroll
    for (int r = 0; r < 8; ++r) {
        int m = m0 + r + half * 8;
        pre[(size_t)m * H_DIM + n] = acc[r] + bias;
    }
}

// ---------------- sequential tanh recurrence --------------------
// Single 1024-thread workgroup (32 waves, one full WGP). h lives in LDS.
#define HP (H_DIM + 8)  // row pad: stride 2064B -> rows land in distinct banks
__global__ void __launch_bounds__(1024)
k_rnn_scan(const float* __restrict__ pre, const bf16_t* __restrict__ Whh,
           const float* __restrict__ b_hh, bf16_t* __restrict__ hs) {
    __shared__ bf16_t h_sh[B_DIM * HP];
    const int tid  = threadIdx.x;
    const int lane = tid & 31;
    const int wave = tid >> 5;
    const int half = lane >> 4;
    const int ln   = lane & 15;
    const int col0 = wave * 32;  // each wave owns 32 h-columns (2 N-tiles)

    for (int i = tid; i < B_DIM * HP; i += 1024) h_sh[i] = f2bf(0.0f);
    __syncthreads();

    const int n0 = col0 + ln;
    const int n1 = col0 + 16 + ln;
    const bf16_t* w0 = Whh + (size_t)n0 * H_DIM;
    const bf16_t* w1 = Whh + (size_t)n1 * H_DIM;
    const float bias0 = b_hh[n0];
    const float bias1 = b_hh[n1];

    for (int t = 0; t < T_DIM; ++t) {
        v8f acc0 = {}, acc1 = {};
        for (int kc = 0; kc < H_DIM; kc += 32) {
            v16bf a  = load_a_split(&h_sh[ln * HP + kc + half * 16]);
            v16bf b0 = load_b_frag(w0 + kc + half * 16);
            v16bf b1 = load_b_frag(w1 + kc + half * 16);
            acc0 = wmma_bf16(a, b0, acc0);
            acc1 = wmma_bf16(a, b1, acc1);
        }
        __syncthreads();  // all waves done reading h before overwrite
#pragma unroll
        for (int r = 0; r < 8; ++r) {
            int bb = r + half * 8;  // batch row 0..15
            size_t prow = ((size_t)bb * T_DIM + t) * H_DIM;
            float f0 = tanhf(acc0[r] + pre[prow + n0] + bias0);
            float f1 = tanhf(acc1[r] + pre[prow + n1] + bias1);
            bf16_t g0 = f2bf(f0), g1 = f2bf(f1);
            h_sh[bb * HP + n0] = g0;
            h_sh[bb * HP + n1] = g1;
            hs[prow + n0] = g0;  // hs[b][t][h] == hs[(b*T+t)*H + h]
            hs[prow + n1] = g1;
        }
        __syncthreads();  // h_new visible before next step's reads
    }
}

// ---------------- FC head: logits = hs @ W_fc^T + b_fc ----------
// Block: 256 threads / 8 waves = 2(M) x 4(N); block tile 64M x 256N.
// Per wave: 32M x 64N (2x4 WMMA accumulators, A reused 4x, B reused 2x).
// B tiles (256 W_fc rows x 32 K) staged to LDS by the Tensor Data Mover,
// double-buffered on TENSORcnt; LDS rows padded 64B->96B via D# pad fields.
#define FC_KCH   32                      // K elems per chunk
#define FC_ROWB  96                      // padded LDS row stride in bytes
#define FC_BUFB  (256 * FC_ROWB)         // 24 KB per buffer
__global__ void __launch_bounds__(256)
k_fc_logits(const bf16_t* __restrict__ hs, const bf16_t* __restrict__ Wfc,
            const float* __restrict__ b_fc, float* __restrict__ logits) {
    __shared__ bf16_t bsh[2 * FC_BUFB / 2];  // 48 KB, two buffers
    const int lane = threadIdx.x & 31;
    const int wave = threadIdx.x >> 5;
    const int half = lane >> 4;
    const int ln   = lane & 15;
    const int mw   = wave & 1;   // 2 M groups of waves
    const int nw   = wave >> 1;  // 4 N groups of waves

    const int m0 = blockIdx.x * 64 + mw * 32;
    const int c0 = blockIdx.y * 256;

    const unsigned lds_base = (unsigned)(size_t)(&bsh[0]);
    const bf16_t* arow0 = hs + (size_t)(m0 + ln) * H_DIM;
    const bf16_t* arow1 = hs + (size_t)(m0 + 16 + ln) * H_DIM;

    v8f acc[2][4] = {};

    const int NI = H_DIM / FC_KCH;  // 32 K chunks
    if (wave == 0) {  // prologue: chunk 0 -> buffer 0
        tdm_load_2d_bf16(lds_base, Wfc + (size_t)c0 * H_DIM, FC_KCH, 256,
                         (unsigned long long)H_DIM,
                         /*pad_enable=*/1, /*interval: 16 dw*/ 3, /*amount: 8 dw*/ 7);
    }
    for (int i = 0; i < NI; ++i) {
        const int p = i & 1;
        __syncthreads();  // previous consumers done with buffer 1-p
        if (wave == 0) {
            if (i + 1 < NI) {
                tdm_load_2d_bf16(lds_base + (unsigned)(1 - p) * FC_BUFB,
                                 Wfc + (size_t)c0 * H_DIM + (size_t)(i + 1) * FC_KCH,
                                 FC_KCH, 256, (unsigned long long)H_DIM, 1, 3, 7);
                __builtin_amdgcn_s_wait_tensorcnt(1);  // chunk i landed
            } else {
                __builtin_amdgcn_s_wait_tensorcnt(0);
            }
        }
        __syncthreads();  // buffer p ready for all waves
        const int kc = i * FC_KCH;
        v16bf a0 = load_a_split(arow0 + kc + half * 16);
        v16bf a1 = load_a_split(arow1 + kc + half * 16);
        const bf16_t* bbase = bsh + (size_t)p * (FC_BUFB / 2);
#pragma unroll
        for (int g = 0; g < 4; ++g) {
            int cl = nw * 64 + g * 16 + ln;  // W_fc row within block tile
            v16bf b = load_b_frag(bbase + cl * (FC_ROWB / 2) + half * 16);
            acc[0][g] = wmma_bf16(a0, b, acc[0][g]);
            acc[1][g] = wmma_bf16(a1, b, acc[1][g]);
        }
    }
#pragma unroll
    for (int mi = 0; mi < 2; ++mi) {
#pragma unroll
        for (int g = 0; g < 4; ++g) {
            int n = c0 + nw * 64 + g * 16 + ln;
            float bias = b_fc[n];
#pragma unroll
            for (int r = 0; r < 8; ++r) {
                int m = m0 + mi * 16 + r + half * 8;
                logits[(size_t)m * V_DIM + n] = acc[mi][g][r] + bias;
            }
        }
    }
}

// ---------------- per-row logsumexp (online softmax) ------------
__global__ void k_row_lse(const float* __restrict__ logits, float* __restrict__ stat) {
    __shared__ float sm[256];
    __shared__ float ss[256];
    const int row = blockIdx.x;
    const float* lr = logits + (size_t)row * V_DIM;
    float m = -INFINITY, s = 0.0f;
    for (int c = threadIdx.x; c < V_DIM; c += 256) {
        float v = lr[c];
        if (v > m) { s = s * __expf(m - v) + 1.0f; m = v; }
        else       { s += __expf(v - m); }
    }
    sm[threadIdx.x] = m;
    ss[threadIdx.x] = s;
    __syncthreads();
    for (int off = 128; off > 0; off >>= 1) {
        if (threadIdx.x < off) {
            float m1 = sm[threadIdx.x], s1 = ss[threadIdx.x];
            float m2 = sm[threadIdx.x + off], s2 = ss[threadIdx.x + off];
            float M = fmaxf(m1, m2);
            sm[threadIdx.x] = M;
            ss[threadIdx.x] = s1 * __expf(m1 - M) + s2 * __expf(m2 - M);
        }
        __syncthreads();
    }
    if (threadIdx.x == 0) stat[row] = sm[0] + __logf(ss[0]);  // row logsumexp
}

// ---------------- NLL + mean reduction --------------------------
__global__ void k_loss(const float* __restrict__ logits, const int* __restrict__ y,
                       const float* __restrict__ stat, float* __restrict__ out_loss) {
    __shared__ float red[256];
    float acc = 0.0f;
    for (int row = threadIdx.x; row < BT_DIM; row += 256) {
        int tgt = y[row];
        acc += stat[row] - logits[(size_t)row * V_DIM + tgt];
    }
    red[threadIdx.x] = acc;
    __syncthreads();
    for (int off = 128; off > 0; off >>= 1) {
        if (threadIdx.x < off) red[threadIdx.x] += red[threadIdx.x + off];
        __syncthreads();
    }
    if (threadIdx.x == 0) out_loss[0] = red[0] / (float)BT_DIM;
}

extern "C" void kernel_launch(void* const* d_in, const int* in_sizes, int n_in,
                              void* d_out, int out_size, void* d_ws, size_t ws_size,
                              hipStream_t stream) {
    const int*   x    = (const int*)d_in[0];
    const int*   y    = (const int*)d_in[1];
    const float* emb  = (const float*)d_in[2];
    const float* W_ih = (const float*)d_in[3];
    const float* b_ih = (const float*)d_in[4];
    const float* W_hh = (const float*)d_in[5];
    const float* b_hh = (const float*)d_in[6];
    const float* W_fc = (const float*)d_in[7];
    const float* b_fc = (const float*)d_in[8];
    (void)in_sizes; (void)n_in; (void)out_size; (void)ws_size;

    float* logits = (float*)d_out;                    // [BT, V]
    float* loss   = logits + (size_t)BT_DIM * V_DIM;  // scalar tail

    char* ws = (char*)d_ws;
    size_t off = 0;
    auto alloc = [&](size_t bytes) -> void* {
        void* p = ws + off;
        off = (off + bytes + 255) & ~(size_t)255;
        return p;
    };
    bf16_t* Wih_bf = (bf16_t*)alloc((size_t)H_DIM * E_DIM * sizeof(bf16_t)); // 1 MB
    bf16_t* Whh_bf = (bf16_t*)alloc((size_t)H_DIM * H_DIM * sizeof(bf16_t)); // 2 MB
    bf16_t* Wfc_bf = (bf16_t*)alloc((size_t)V_DIM * H_DIM * sizeof(bf16_t)); // 65.5 MB (L2)
    float*  pre    = (float*) alloc((size_t)BT_DIM * H_DIM * sizeof(float)); // 16 MB
    bf16_t* hs     = (bf16_t*)alloc((size_t)BT_DIM * H_DIM * sizeof(bf16_t));// 8 MB
    float*  stat   = (float*) alloc((size_t)BT_DIM * sizeof(float));         // 16 KB

    k_f32_to_bf16<<<1024, 256, 0, stream>>>(W_ih, Wih_bf, (long)H_DIM * E_DIM);
    k_f32_to_bf16<<<2048, 256, 0, stream>>>(W_hh, Whh_bf, (long)H_DIM * H_DIM);
    k_f32_to_bf16<<<8192, 256, 0, stream>>>(W_fc, Wfc_bf, (long)V_DIM * H_DIM);

    k_embed_proj<<<dim3(BT_DIM / 16, H_DIM / 128), 256, 0, stream>>>(x, emb, Wih_bf,
                                                                     b_ih, pre);
    k_rnn_scan<<<1, 1024, 0, stream>>>(pre, Whh_bf, b_hh, hs);
    k_fc_logits<<<dim3(BT_DIM / 64, V_DIM / 256), 256, 0, stream>>>(hs, Wfc_bf, b_fc,
                                                                    logits);
    k_row_lse<<<BT_DIM, 256, 0, stream>>>(logits, stat);
    k_loss<<<1, 256, 0, stream>>>(logits, y, stat, loss);
}
